// Messagepasstoedge_1632087573107
// MI455X (gfx1250) — matrix operand order repaired and checked by reading, not compiled, not run on hardware
//
#include <hip/hip_runtime.h>
#include <math.h>

// ---------------------------------------------------------------------------
// Problem constants (from reference): N=50000 nodes, E=100000 edges,
// D=512 model dim, DH=2048 hidden dim.  Output = [feats (N*D) | edge_out (E*D)]
// ---------------------------------------------------------------------------
#define NN    50000
#define EE    100000
#define DD    512
#define DHH   2048

typedef __attribute__((ext_vector_type(16))) __bf16 v16bf;
typedef __attribute__((ext_vector_type(8)))  __bf16 v8bf;
typedef __attribute__((ext_vector_type(4)))  __bf16 v4bf;
typedef __attribute__((ext_vector_type(8)))  float  v8f;
typedef __attribute__((ext_vector_type(4)))  float  v4f;

__device__ __forceinline__ float gelu_f(float x) {
    return 0.5f * x * (1.0f + erff(x * 0.70710678118654752440f));
}

// block-wide sum over 256 threads
__device__ __forceinline__ float blk_sum(float v, float* s) {
    const int t = threadIdx.x;
    s[t] = v; __syncthreads();
    #pragma unroll
    for (int off = 128; off > 0; off >>= 1) {
        if (t < off) s[t] += s[t + off];
        __syncthreads();
    }
    float r = s[0]; __syncthreads();
    return r;
}

// ---------------------------------------------------------------------------
// bf16 WMMA GEMM:  C[M x Ncols] = act(A[M x K] * W[Ncols x K]^T + bias)
//   TA   : float (converted to bf16 at LDS-store time) or __bf16
//   MODE : 0 = store fp32 ; 1 = relu -> store bf16 ; 2 = gelu -> store bf16
// Block tile 128x128, K-step 64, register-staged double-buffered LDS.
// 8 waves of 32 lanes; each wave: 32x64 tile = 2x4 accumulators, issuing
// 16 v_wmma_f32_16x16x32_bf16 per K-iteration.
// ---------------------------------------------------------------------------
#define BM  128
#define BN  128
#define BK  64
#define LDT 72   // LDS row stride in bf16 elements (64 data + 8 pad, 144 B)

// raw per-thread staging loads (branch-free; rows clamped by caller)
__device__ __forceinline__ v4f  load_a_raw(const float*  p) { return *(const v4f*)p;  }
__device__ __forceinline__ v4bf load_a_raw(const __bf16* p) { return *(const v4bf*)p; }
__device__ __forceinline__ v4bf to_bf16x4(v4f f) {
    v4bf h;
    h[0] = (__bf16)f[0]; h[1] = (__bf16)f[1];
    h[2] = (__bf16)f[2]; h[3] = (__bf16)f[3];
    return h;
}
__device__ __forceinline__ v4bf to_bf16x4(v4bf h) { return h; }

template <typename TA, int MODE>
__global__ __launch_bounds__(256) void gemm_bf16_kernel(
    const TA* __restrict__ A, const __bf16* __restrict__ W,
    const float* __restrict__ bias,
    float* __restrict__ Cf, __bf16* __restrict__ Cb,
    int M, int K, int Ncols)
{
    __shared__ __bf16 lds_a[2][BM * LDT];
    __shared__ __bf16 lds_b[2][BN * LDT];

    const int tid  = threadIdx.x;
    const int bm   = blockIdx.x * BM;
    const int bn   = blockIdx.y * BN;
    const int wave = tid >> 5;
    const int lane = tid & 31;
    const int wm   = wave & 3;   // 4 waves along M (32 rows each)
    const int wn   = wave >> 2;  // 2 waves along N (64 cols each)
    const int hf   = lane >> 4;  // half-wave (0: lanes 0-15, 1: lanes 16-31)
    const int l16  = lane & 15;

    using AReg = decltype(load_a_raw((const TA*)0));
    AReg areg[8];   // A tile: 2048 4-elt chunks / 256 threads
    v8bf breg[4];   // B tile: 1024 8-elt chunks / 256 threads

    // branch-free global -> register staging for K-slice starting at k0
    auto stage_load = [&](int k0) {
        #pragma unroll
        for (int it = 0; it < 8; ++it) {
            const int fi  = it * 256 + tid;
            int grow = bm + (fi >> 4);           // row 0..127
            grow = grow < M ? grow : M - 1;      // clamp: junk rows only feed junk C rows
            areg[it] = load_a_raw(A + (size_t)grow * K + k0 + (fi & 15) * 4);
        }
        #pragma unroll
        for (int it = 0; it < 4; ++it) {
            const int vi = it * 256 + tid;
            breg[it] = *(const v8bf*)(W + (size_t)(bn + (vi >> 3)) * K + k0 + (vi & 7) * 8);
        }
    };
    auto stage_store = [&](int buf) {
        #pragma unroll
        for (int it = 0; it < 8; ++it) {
            const int fi = it * 256 + tid;
            *(v4bf*)(&lds_a[buf][(fi >> 4) * LDT + (fi & 15) * 4]) = to_bf16x4(areg[it]);
        }
        #pragma unroll
        for (int it = 0; it < 4; ++it) {
            const int vi = it * 256 + tid;
            *(v8bf*)(&lds_b[buf][(vi >> 3) * LDT + (vi & 7) * 8]) = breg[it];
        }
    };

    v8f acc[2][4];
    #pragma unroll
    for (int i = 0; i < 2; ++i)
        #pragma unroll
        for (int j = 0; j < 4; ++j)
            #pragma unroll
            for (int r = 0; r < 8; ++r) acc[i][j][r] = 0.0f;

    const int nk = K / BK;
    stage_load(0);
    stage_store(0);
    __syncthreads();

    for (int kt = 0; kt < nk; ++kt) {
        const int cur = kt & 1;
        // issue next tile's global loads before computing (latency hidden by WMMA)
        if (kt + 1 < nk) stage_load((kt + 1) * BK);
        if (kt + 2 < nk)  // L2 prefetch of the tile after next (weights)
            __builtin_prefetch((const void*)(W + (size_t)(bn + (tid >> 1)) * K + (kt + 2) * BK), 0, 3);

        // ---- compute: two K=32 sub-steps from current LDS buffer
        #pragma unroll
        for (int ks = 0; ks < 2; ++ks) {
            v16bf afrag[2], bfrag[4];
            #pragma unroll
            for (int mt = 0; mt < 2; ++mt) {
                // A 16x32 layout: lane l16 = row M; half 0: K 0-7 & 16-23, half 1: K 8-15 & 24-31
                const __bf16* ap = &lds_a[cur][(wm * 32 + mt * 16 + l16) * LDT + ks * 32 + hf * 8];
                v8bf lo = *(const v8bf*)ap;
                v8bf hi = *(const v8bf*)(ap + 16);
                afrag[mt] = __builtin_shufflevector(lo, hi, 0,1,2,3,4,5,6,7,8,9,10,11,12,13,14,15);
            }
            #pragma unroll
            for (int nt = 0; nt < 4; ++nt) {
                // B 32x16 layout: lane l16 = col N; half 0: K 0-7 & 16-23, half 1: K 8-15 & 24-31
                const __bf16* bp = &lds_b[cur][(wn * 64 + nt * 16 + l16) * LDT + ks * 32 + hf * 8];
                v8bf lo = *(const v8bf*)bp;
                v8bf hi = *(const v8bf*)(bp + 16);
                bfrag[nt] = __builtin_shufflevector(lo, hi, 0,1,2,3,4,5,6,7,8,9,10,11,12,13,14,15);
            }
            #pragma unroll
            for (int mt = 0; mt < 2; ++mt)
                #pragma unroll
                for (int nt = 0; nt < 4; ++nt)
                    acc[mt][nt] = __builtin_amdgcn_wmma_f32_16x16x32_bf16(
                        false, afrag[mt], false, bfrag[nt],
                        (short)0, acc[mt][nt], false, false);
        }

        // write next tile into the alternate buffer; one barrier per iteration
        if (kt + 1 < nk) stage_store(cur ^ 1);
        __syncthreads();
    }

    // ---- epilogue: C/D layout — lane l16 = N col, VGPR r = row (r + 8*half)
    #pragma unroll
    for (int mt = 0; mt < 2; ++mt) {
        #pragma unroll
        for (int nt = 0; nt < 4; ++nt) {
            const int col = bn + wn * 64 + nt * 16 + l16;
            const float bv = bias ? bias[col] : 0.0f;
            #pragma unroll
            for (int r = 0; r < 8; ++r) {
                const int row = bm + wm * 32 + mt * 16 + hf * 8 + r;
                if (row < M) {
                    float val = acc[mt][nt][r] + bv;
                    if (MODE == 1) {
                        val = fmaxf(val, 0.0f);
                        Cb[(size_t)row * Ncols + col] = (__bf16)val;
                    } else if (MODE == 2) {
                        val = gelu_f(val);
                        Cb[(size_t)row * Ncols + col] = (__bf16)val;
                    } else {
                        Cf[(size_t)row * Ncols + col] = val;
                    }
                }
            }
        }
    }
}

// ---------------------------------------------------------------------------
// fp32 -> bf16 weight conversion
// ---------------------------------------------------------------------------
__global__ void f32_to_bf16_kernel(const float* __restrict__ s,
                                   __bf16* __restrict__ d, int n) {
    int i = blockIdx.x * 256 + threadIdx.x;
    if (i < n) d[i] = (__bf16)s[i];
}

// ---------------------------------------------------------------------------
// kv gather: kv[2e] = ef[row(e)], kv[2e+1] = ef[col(e)]   (bf16, D=512)
// ---------------------------------------------------------------------------
__global__ __launch_bounds__(128) void gather_kv_kernel(
    const __bf16* __restrict__ ef, const int* __restrict__ eidx,
    __bf16* __restrict__ kv) {
    const int e = blockIdx.x;
    const int t = threadIdx.x;           // 128 threads, 4 bf16 each
    const int r = eidx[2 * e + 0];
    const int c = eidx[2 * e + 1];
    const v4bf* s0 = (const v4bf*)(ef + (size_t)r * DD);
    const v4bf* s1 = (const v4bf*)(ef + (size_t)c * DD);
    v4bf* d0 = (v4bf*)(kv + (size_t)(2 * e) * DD);
    v4bf* d1 = d0 + (DD / 4);
    d0[t] = s0[t];
    d1[t] = s1[t];
}

// ---------------------------------------------------------------------------
// attention + residual + gelu + layernorm:   x = LN(gelu(att + edge_attr))
// one 256-thread block per edge, 2 elements/thread over D=512
// ---------------------------------------------------------------------------
__global__ __launch_bounds__(256) void attn_kernel(
    const float* __restrict__ q, const float* __restrict__ k,
    const float* __restrict__ v, const float* __restrict__ ea,
    const float* __restrict__ gn, const float* __restrict__ bn,
    float* __restrict__ xf, __bf16* __restrict__ xb) {
    __shared__ float sbuf[256];
    __shared__ float sw[2];
    const int e = blockIdx.x;
    const int t = threadIdx.x;
    const float* qe = q + (size_t)e * DD;
    const float* k0 = k + (size_t)(2 * e) * DD;
    const float* k1 = k0 + DD;
    const float* v0 = v + (size_t)(2 * e) * DD;
    const float* v1 = v0 + DD;
    const float* ee = ea + (size_t)e * DD;

    const float qa = qe[t], qb = qe[t + 256];
    float s0 = qa * k0[t] + qb * k0[t + 256];
    float s1 = qa * k1[t] + qb * k1[t + 256];
    s0 = blk_sum(s0, sbuf);
    s1 = blk_sum(s1, sbuf);
    if (t == 0) {
        const float scale = 0.044194173824159216f;  // 1/sqrt(512)
        float a0 = s0 * scale, a1 = s1 * scale;
        float m = fmaxf(a0, a1);
        float e0 = __expf(a0 - m), e1 = __expf(a1 - m);
        float inv = 1.0f / (e0 + e1);
        sw[0] = e0 * inv; sw[1] = e1 * inv;
    }
    __syncthreads();
    const float w0 = sw[0], w1 = sw[1];

    float g0 = gelu_f(w0 * v0[t]       + w1 * v1[t]       + ee[t]);
    float g1 = gelu_f(w0 * v0[t + 256] + w1 * v1[t + 256] + ee[t + 256]);
    const float mean = blk_sum(g0 + g1, sbuf) * (1.0f / (float)DD);
    const float d0 = g0 - mean, d1 = g1 - mean;
    const float var = blk_sum(d0 * d0 + d1 * d1, sbuf) * (1.0f / (float)DD);
    const float rstd = rsqrtf(var + 1e-5f);
    const float o0 = d0 * rstd * gn[t]       + bn[t];
    const float o1 = d1 * rstd * gn[t + 256] + bn[t + 256];
    const size_t base = (size_t)e * DD;
    xf[base + t] = o0;       xf[base + t + 256] = o1;
    xb[base + t] = (__bf16)o0; xb[base + t + 256] = (__bf16)o1;
}

// ---------------------------------------------------------------------------
// in-place LayerNorm over DH=2048 on bf16 rows (input already gelu'd)
// ---------------------------------------------------------------------------
__global__ __launch_bounds__(256) void ln_bf16_kernel(
    __bf16* __restrict__ h, const float* __restrict__ g,
    const float* __restrict__ b) {
    __shared__ float sbuf[256];
    const int r = blockIdx.x;
    const int t = threadIdx.x;
    __bf16* hr = h + (size_t)r * DHH;
    float x[8];
    float s = 0.0f;
    #pragma unroll
    for (int i = 0; i < 8; ++i) { x[i] = (float)hr[t + i * 256]; s += x[i]; }
    const float mean = blk_sum(s, sbuf) * (1.0f / (float)DHH);
    float ss = 0.0f;
    #pragma unroll
    for (int i = 0; i < 8; ++i) { float d = x[i] - mean; ss += d * d; }
    const float var = blk_sum(ss, sbuf) * (1.0f / (float)DHH);
    const float rstd = rsqrtf(var + 1e-5f);
    #pragma unroll
    for (int i = 0; i < 8; ++i) {
        const int c = t + i * 256;
        hr[c] = (__bf16)((x[i] - mean) * rstd * g[c] + b[c]);
    }
}

// ---------------------------------------------------------------------------
// final: edge_out = LN(gelu(y + x)), fp32 out
// ---------------------------------------------------------------------------
__global__ __launch_bounds__(256) void final_kernel(
    const float* __restrict__ y, const float* __restrict__ x,
    const float* __restrict__ gn, const float* __restrict__ bn,
    float* __restrict__ out) {
    __shared__ float sbuf[256];
    const int e = blockIdx.x;
    const int t = threadIdx.x;
    const size_t base = (size_t)e * DD;
    float g0 = gelu_f(y[base + t]       + x[base + t]);
    float g1 = gelu_f(y[base + t + 256] + x[base + t + 256]);
    const float mean = blk_sum(g0 + g1, sbuf) * (1.0f / (float)DD);
    const float d0 = g0 - mean, d1 = g1 - mean;
    const float var = blk_sum(d0 * d0 + d1 * d1, sbuf) * (1.0f / (float)DD);
    const float rstd = rsqrtf(var + 1e-5f);
    out[base + t]       = d0 * rstd * gn[t]       + bn[t];
    out[base + t + 256] = d1 * rstd * gn[t + 256] + bn[t + 256];
}

// ---------------------------------------------------------------------------
// host-side orchestration
// ---------------------------------------------------------------------------
extern "C" void kernel_launch(void* const* d_in, const int* in_sizes, int n_in,
                              void* d_out, int out_size, void* d_ws, size_t ws_size,
                              hipStream_t stream) {
    (void)in_sizes; (void)n_in; (void)out_size; (void)ws_size;
    const float* feats     = (const float*)d_in[0];
    const int*   edge_idx  = (const int*)  d_in[1];
    const float* edge_attr = (const float*)d_in[2];
    const float* W_ne = (const float*)d_in[3];   const float* b_ne = (const float*)d_in[4];
    const float* W_q  = (const float*)d_in[5];   const float* b_q  = (const float*)d_in[6];
    const float* W_k  = (const float*)d_in[7];   const float* b_k  = (const float*)d_in[8];
    const float* W_v  = (const float*)d_in[9];   const float* b_v  = (const float*)d_in[10];
    const float* g_norm  = (const float*)d_in[11]; const float* bt_norm = (const float*)d_in[12];
    const float* W1  = (const float*)d_in[13];   const float* b1  = (const float*)d_in[14];
    const float* g_ln2 = (const float*)d_in[15]; const float* bt_ln2 = (const float*)d_in[16];
    const float* W2  = (const float*)d_in[17];   const float* b2  = (const float*)d_in[18];

    char* ws = (char*)d_ws;
    // --- workspace layout (bytes); peak use ~1.60 GB with region reuse ---
    const size_t o_wne = 0;                                   // 512x512 bf16
    const size_t o_wq  = o_wne + (size_t)DD * DD * 2;
    const size_t o_wk  = o_wq  + (size_t)DD * DD * 2;
    const size_t o_wv  = o_wk  + (size_t)DD * DD * 2;
    const size_t o_w1  = o_wv  + (size_t)DD * DD * 2;         // 2048x512 bf16
    const size_t o_w2  = o_w1  + (size_t)DHH * DD * 2;        // 512x2048 bf16
    const size_t o_ef  = o_w2  + (size_t)DD * DHH * 2;        // ef bf16   N*D
    const size_t o_kv  = o_ef  + (size_t)NN * DD * 2;         // kv bf16  2E*D
    const size_t o_q   = o_kv  + (size_t)2 * EE * DD * 2;     // q  f32    E*D
    const size_t o_k   = o_q   + (size_t)EE * DD * 4;         // k  f32   2E*D
    const size_t o_v   = o_k   + (size_t)2 * EE * DD * 4;     // v  f32   2E*D
    const size_t o_x   = o_v   + (size_t)2 * EE * DD * 4;     // x  f32    E*D
    const size_t o_xb  = o_x   + (size_t)EE * DD * 4;         // x  bf16   E*D
    // after attention, q/k/v are dead -> reuse their region:
    const size_t o_hg  = o_q;                                 // gelu(h) bf16 E*DH
    const size_t o_y   = o_hg + (size_t)EE * DHH * 2;         // y f32 E*D

    __bf16* wb_ne = (__bf16*)(ws + o_wne);
    __bf16* wb_q  = (__bf16*)(ws + o_wq);
    __bf16* wb_k  = (__bf16*)(ws + o_wk);
    __bf16* wb_v  = (__bf16*)(ws + o_wv);
    __bf16* wb_1  = (__bf16*)(ws + o_w1);
    __bf16* wb_2  = (__bf16*)(ws + o_w2);
    __bf16* ef_b  = (__bf16*)(ws + o_ef);
    __bf16* kv_b  = (__bf16*)(ws + o_kv);
    float*  q_f   = (float*) (ws + o_q);
    float*  k_f   = (float*) (ws + o_k);
    float*  v_f   = (float*) (ws + o_v);
    float*  x_f   = (float*) (ws + o_x);
    __bf16* x_b   = (__bf16*)(ws + o_xb);
    __bf16* h_b   = (__bf16*)(ws + o_hg);
    float*  y_f   = (float*) (ws + o_y);

    float* out_feats = (float*)d_out;
    float* out_edge  = out_feats + (size_t)NN * DD;

    // 0) pass feats through to output (graph-capture safe d2d async copy)
    hipMemcpyAsync(out_feats, feats, (size_t)NN * DD * sizeof(float),
                   hipMemcpyDeviceToDevice, stream);

    // 1) weights -> bf16
    {
        const int nA = DD * DD, nB = DHH * DD;
        f32_to_bf16_kernel<<<(nA + 255) / 256, 256, 0, stream>>>(W_ne, wb_ne, nA);
        f32_to_bf16_kernel<<<(nA + 255) / 256, 256, 0, stream>>>(W_q,  wb_q,  nA);
        f32_to_bf16_kernel<<<(nA + 255) / 256, 256, 0, stream>>>(W_k,  wb_k,  nA);
        f32_to_bf16_kernel<<<(nA + 255) / 256, 256, 0, stream>>>(W_v,  wb_v,  nA);
        f32_to_bf16_kernel<<<(nB + 255) / 256, 256, 0, stream>>>(W1,   wb_1,  nB);
        f32_to_bf16_kernel<<<(nB + 255) / 256, 256, 0, stream>>>(W2,   wb_2,  nB);
    }

    const dim3 blk(256);
    // 2) ef = relu(feats @ W_ne^T + b_ne)  -> bf16
    gemm_bf16_kernel<float, 1><<<dim3((NN + BM - 1) / BM, DD / BN), blk, 0, stream>>>(
        feats, wb_ne, b_ne, nullptr, ef_b, NN, DD, DD);
    // 3) gather kv_in rows
    gather_kv_kernel<<<dim3(EE), dim3(128), 0, stream>>>(ef_b, edge_idx, kv_b);
    // 4) q = edge_attr @ W_q^T + b_q  -> fp32
    gemm_bf16_kernel<float, 0><<<dim3((EE + BM - 1) / BM, DD / BN), blk, 0, stream>>>(
        edge_attr, wb_q, b_q, q_f, nullptr, EE, DD, DD);
    // 5) k = kv @ W_k^T + b_k ; v = kv @ W_v^T + b_v  -> fp32
    gemm_bf16_kernel<__bf16, 0><<<dim3((2 * EE + BM - 1) / BM, DD / BN), blk, 0, stream>>>(
        kv_b, wb_k, b_k, k_f, nullptr, 2 * EE, DD, DD);
    gemm_bf16_kernel<__bf16, 0><<<dim3((2 * EE + BM - 1) / BM, DD / BN), blk, 0, stream>>>(
        kv_b, wb_v, b_v, v_f, nullptr, 2 * EE, DD, DD);
    // 6) attention + softmax + residual gelu + LN  -> x (f32 + bf16)
    attn_kernel<<<dim3(EE), blk, 0, stream>>>(q_f, k_f, v_f, edge_attr,
                                              g_norm, bt_norm, x_f, x_b);
    // 7) h_pre = gelu(x @ W1^T + b1) -> bf16 (LN applied next, in place)
    gemm_bf16_kernel<__bf16, 2><<<dim3((EE + BM - 1) / BM, DHH / BN), blk, 0, stream>>>(
        x_b, wb_1, b1, nullptr, h_b, EE, DD, DHH);
    // 8) h = LN(h_pre) in place over DH
    ln_bf16_kernel<<<dim3(EE), blk, 0, stream>>>(h_b, g_ln2, bt_ln2);
    // 9) y = h @ W2^T + b2 -> fp32
    gemm_bf16_kernel<__bf16, 0><<<dim3((EE + BM - 1) / BM, DD / BN), blk, 0, stream>>>(
        h_b, wb_2, b2, y_f, nullptr, EE, DHH, DD);
    // 10) edge_out = LN(gelu(y + x))
    final_kernel<<<dim3(EE), blk, 0, stream>>>(y_f, x_f, g_norm, bt_norm, out_edge);
}
